// AttentiveGatingv2_58308476011153
// MI455X (gfx1250) — compile-verified, matrix operand order checked by reading, not run on hardware
//
#include <hip/hip_runtime.h>
#include <hip/hip_bf16.h>

typedef __attribute__((ext_vector_type(16))) _Float16 v16h;
typedef __attribute__((ext_vector_type(8)))  _Float16 v8h;
typedef __attribute__((ext_vector_type(8)))  float    v8f;

#define D_MODEL   32
#define N_HEADS   4
#define HEAD_DIM  8
#define N_EXPERTS 8
#define K_STEPS   6
#define IN_CH     64
#define BATCH     64
#define T_LEN     96
#define N_TOK     207

// Block: 192 threads = 6 waves. Each wave owns a 16-row tile of the flattened
// M = BATCH*N_TOK*K_STEPS = 79488 row dimension. Block covers 96 rows = 16 sites.
// Grid = 79488/96 = 828 blocks (exact, no tail -> EXEC all ones for WMMA).

// Load a 16x32-f16 WMMA operand fragment (A or B, they stripe identically) from a
// row-major [row][rowStride] f16 array. Per ISA: lanes 0-15 hold K=0..7 & 16..23,
// lanes 16-31 hold K=8..15 & 24..31, as contiguous 8-half (16 B) groups.
__device__ __forceinline__ v16h ld_frag(const _Float16* base, int rowStride,
                                        int row, int kbase) {
  const _Float16* p = base + row * rowStride + kbase;
  v8h lo = *(const v8h*)(p);
  v8h hi = *(const v8h*)(p + 16);
  v16h a;
#pragma unroll
  for (int i = 0; i < 8; ++i) { a[i] = lo[i]; a[8 + i] = hi[i]; }
  return a;
}

__device__ __forceinline__ v16h cvt_frag(float4 f0, float4 f1, float4 f2, float4 f3) {
  v16h a;
  a[0]  = (_Float16)f0.x; a[1]  = (_Float16)f0.y; a[2]  = (_Float16)f0.z; a[3]  = (_Float16)f0.w;
  a[4]  = (_Float16)f1.x; a[5]  = (_Float16)f1.y; a[6]  = (_Float16)f1.z; a[7]  = (_Float16)f1.w;
  a[8]  = (_Float16)f2.x; a[9]  = (_Float16)f2.y; a[10] = (_Float16)f2.z; a[11] = (_Float16)f2.w;
  a[12] = (_Float16)f3.x; a[13] = (_Float16)f3.y; a[14] = (_Float16)f3.z; a[15] = (_Float16)f3.w;
  return a;
}

__global__ __launch_bounds__(192)
void attentive_gating_kernel(const float* __restrict__ x,
                             const float* __restrict__ proj_w,
                             const float* __restrict__ proj_b,
                             const float* __restrict__ in_w,
                             const float* __restrict__ in_b,
                             const float* __restrict__ out_w,
                             const float* __restrict__ out_b,
                             const float* __restrict__ fc_w,
                             const float* __restrict__ fc_b,
                             float* __restrict__ out) {
  // f16 weight tiles, stored n-major so B-fragments are contiguous-K per lane.
  __shared__ alignas(16) _Float16 sW1[32 * 64];   // proj_w  (d=32 rows, C=64)
  __shared__ alignas(16) _Float16 sW2[96 * 32];   // in_proj_w (96 rows, 32)
  __shared__ alignas(16) _Float16 sW3[32 * 32];   // out_w (32 rows, 32)
  __shared__ alignas(16) _Float16 sQKV[96 * 96];  // per-block qkv rows
  __shared__ alignas(16) _Float16 sCTX[96 * 32];  // attention context
  __shared__ alignas(16) float    sH[96 * 32];    // out-proj result (f32)
  __shared__ alignas(16) _Float16 sZ[6][16 * 32]; // per-wave z scratch (transpose hop)

  const int tid  = threadIdx.x;
  const int wave = tid >> 5;
  const int lane = tid & 31;
  const int lm   = lane & 15;           // row (A) / col (B) within 16-tile
  const int koff = (lane >> 4) * 8;     // K sub-offset per ISA striping
  const int mrow = (lane >> 4) * 8;     // accumulator row group (C/D layout)

  // ---- Issue the x loads FIRST: they depend only on global memory, so their
  // HBM latency overlaps the weight-conversion phase and the barrier below.
  const int r    = blockIdx.x * 96 + wave * 16 + lm;
  const int site = r / 6;
  const int kk   = r - site * 6;
  const int b    = site / N_TOK;
  const int n    = site - b * N_TOK;
  const float* xr = x + ((((size_t)b * T_LEN) + (T_LEN - K_STEPS + kk)) * N_TOK + n) * IN_CH;
  float4 xf[8];
#pragma unroll
  for (int t = 0; t < 2; ++t) {
    const int c0 = t * 32 + koff;
    xf[t * 4 + 0] = *(const float4*)(xr + c0);
    xf[t * 4 + 1] = *(const float4*)(xr + c0 + 4);
    xf[t * 4 + 2] = *(const float4*)(xr + c0 + 16);
    xf[t * 4 + 3] = *(const float4*)(xr + c0 + 20);
  }

  // ---- Phase 0: weights f32 -> f16 into LDS (overlaps x loads in flight) --
  for (int i = tid; i < 32 * 64; i += 192) sW1[i] = (_Float16)proj_w[i];
  for (int i = tid; i < 96 * 32; i += 192) sW2[i] = (_Float16)in_w[i];
  for (int i = tid; i < 32 * 32; i += 192) sW3[i] = (_Float16)out_w[i];
  __syncthreads();

  // ---- Phase 1: GEMM1 (proj, K=64) + GEMM2 (qkv, K=32) per wave -----------
  {
    v16h a0 = cvt_frag(xf[0], xf[1], xf[2], xf[3]);
    v16h a1 = cvt_frag(xf[4], xf[5], xf[6], xf[7]);

    // GEMM1: z[16x32] = A[16x64] * W1^T, two N-tiles, two K-steps each.
    _Float16* z = sZ[wave];
#pragma unroll
    for (int t = 0; t < 2; ++t) {
      v8f c = {};
      v16h b0 = ld_frag(sW1, 64, t * 16 + lm, 0 + koff);
      c = __builtin_amdgcn_wmma_f32_16x16x32_f16(false, a0, false, b0, (short)0, c, false, false);
      v16h b1 = ld_frag(sW1, 64, t * 16 + lm, 32 + koff);
      c = __builtin_amdgcn_wmma_f32_16x16x32_f16(false, a1, false, b1, (short)0, c, false, false);
      const float bias = proj_b[t * 16 + lm];
#pragma unroll
      for (int v = 0; v < 8; ++v)
        z[(mrow + v) * 32 + t * 16 + lm] = (_Float16)(c[v] + bias);
    }

    // Reload z as an A fragment (LDS hop performs the lane<->row transpose).
    v16h a2 = ld_frag(z, 32, lm, koff);

    // GEMM2: qkv[16x96] = z[16x32] * W2^T, six N-tiles.
#pragma unroll
    for (int t = 0; t < 6; ++t) {
      v16h bw = ld_frag(sW2, 32, t * 16 + lm, koff);
      v8f c = {};
      c = __builtin_amdgcn_wmma_f32_16x16x32_f16(false, a2, false, bw, (short)0, c, false, false);
      const float bias = in_b[t * 16 + lm];
      const int row0 = wave * 16 + mrow;
#pragma unroll
      for (int v = 0; v < 8; ++v)
        sQKV[(row0 + v) * 96 + t * 16 + lm] = (_Float16)(c[v] + bias);
    }
  }
  __syncthreads();

  // ---- Phase 2: attention on VALU, all 192 threads busy -------------------
  // 16 sites x 4 heads x 3 q-row-pairs = 384 row-softmaxes, 2 per thread.
  {
    const int s  = tid / 12;           // site within block (0..15)
    const int w  = tid - s * 12;       // 0..11
    const int h  = w / 3;              // head (0..3)
    const int qp = w - h * 3;          // q-row pair (0..2)
    const int r0 = s * 6;
    float k[6][8], vv[6][8];
#pragma unroll
    for (int j = 0; j < 6; ++j)
#pragma unroll
      for (int d = 0; d < 8; ++d) {
        k[j][d]  = (float)sQKV[(r0 + j) * 96 + 32 + h * 8 + d];
        vv[j][d] = (float)sQKV[(r0 + j) * 96 + 64 + h * 8 + d];
      }
    const float scale = 0.35355339059327373f;  // 1/sqrt(8)
#pragma unroll
    for (int ii = 0; ii < 2; ++ii) {
      const int i = qp * 2 + ii;
      float q[8];
#pragma unroll
      for (int d = 0; d < 8; ++d) q[d] = (float)sQKV[(r0 + i) * 96 + h * 8 + d];
      float sc[6], mx = -1e30f;
#pragma unroll
      for (int j = 0; j < 6; ++j) {
        float dt = 0.f;
#pragma unroll
        for (int d = 0; d < 8; ++d) dt += q[d] * k[j][d];
        sc[j] = dt * scale;
        mx = fmaxf(mx, sc[j]);
      }
      float sum = 0.f;
#pragma unroll
      for (int j = 0; j < 6; ++j) { sc[j] = __expf(sc[j] - mx); sum += sc[j]; }
      const float inv = 1.0f / sum;
#pragma unroll
      for (int d = 0; d < 8; ++d) {
        float cx = 0.f;
#pragma unroll
        for (int j = 0; j < 6; ++j) cx += sc[j] * vv[j][d];
        sCTX[(r0 + i) * 32 + h * 8 + d] = (_Float16)(cx * inv);
      }
    }
  }
  __syncthreads();

  // ---- Phase 3: GEMM3 (out proj, K=32) per wave ---------------------------
  {
    v16h a3 = ld_frag(sCTX + wave * 16 * 32, 32, lm, koff);
#pragma unroll
    for (int t = 0; t < 2; ++t) {
      v16h bw = ld_frag(sW3, 32, t * 16 + lm, koff);
      v8f c = {};
      c = __builtin_amdgcn_wmma_f32_16x16x32_f16(false, a3, false, bw, (short)0, c, false, false);
      const float bias = out_b[t * 16 + lm];
      const int row0 = wave * 16 + mrow;
#pragma unroll
      for (int v = 0; v < 8; ++v)
        sH[(row0 + v) * 32 + t * 16 + lm] = c[v] + bias;
    }
  }
  __syncthreads();

  // ---- Phase 4: mean over K_STEPS, fc head, softmax -----------------------
  // One thread per (site, expert) = 128 threads; 8-expert softmax reduced with
  // quarter-wave shuffles (wave32 lane permutes).
  if (tid < 128) {
    const int s = tid >> 3;     // site (0..15)
    const int e = tid & 7;      // expert (0..7)
    float logit = fc_b[e];
#pragma unroll
    for (int c = 0; c < 32; ++c) {
      float acc = 0.f;
#pragma unroll
      for (int i = 0; i < 6; ++i) acc += sH[(s * 6 + i) * 32 + c];
      logit += (acc * (1.0f / 6.0f)) * fc_w[e * 32 + c];
    }
    // max over the 8 experts (lanes e..e^7 within the same quarter-wave)
    float mx = logit;
#pragma unroll
    for (int m = 1; m < 8; m <<= 1) mx = fmaxf(mx, __shfl_xor(mx, m, 8));
    float ex = __expf(logit - mx);
    float sum = ex;
#pragma unroll
    for (int m = 1; m < 8; m <<= 1) sum += __shfl_xor(sum, m, 8);
    out[((size_t)blockIdx.x * 16 + s) * 8 + e] = ex / sum;
  }
}

extern "C" void kernel_launch(void* const* d_in, const int* in_sizes, int n_in,
                              void* d_out, int out_size, void* d_ws, size_t ws_size,
                              hipStream_t stream) {
  (void)in_sizes; (void)n_in; (void)out_size; (void)d_ws; (void)ws_size;
  const float* x      = (const float*)d_in[0];
  const float* proj_w = (const float*)d_in[1];
  const float* proj_b = (const float*)d_in[2];
  const float* in_w   = (const float*)d_in[3];
  const float* in_b   = (const float*)d_in[4];
  const float* out_w  = (const float*)d_in[5];
  const float* out_b  = (const float*)d_in[6];
  const float* fc_w   = (const float*)d_in[7];
  const float* fc_b   = (const float*)d_in[8];
  float* out = (float*)d_out;

  const int sites  = BATCH * N_TOK;          // 13248
  const int blocks = sites / 16;             // 828 (exact)
  attentive_gating_kernel<<<blocks, 192, 0, stream>>>(
      x, proj_w, proj_b, in_w, in_b, out_w, out_b, fc_w, fc_b, out);
}
